// WaveRNN_18408229831248
// MI455X (gfx1250) — matrix operand order repaired and checked by reading, not compiled, not run on hardware
//
#include <hip/hip_runtime.h>
#include <hip/hip_bf16.h>
#include <math.h>

typedef __attribute__((ext_vector_type(16))) _Float16 v16h;
typedef __attribute__((ext_vector_type(8)))  _Float16 v8h;
typedef __attribute__((ext_vector_type(8)))  float    v8f;

#define HS 384   // hidden
#define SS 192   // split
#define QQ 256   // quantization

// ---------- fragment loaders (contiguous k order per lane half, same for A and B) ----------
// A 16x32 f16: lane = 16*g + m ; halves 0..15 -> K = k0+g*16 .. +15
__device__ __forceinline__ v16h load_fragA(const _Float16* __restrict__ p) {
  v16h r;
  ((v8h*)&r)[0] = *(const v8h*)(p);
  ((v8h*)&r)[1] = *(const v8h*)(p + 8);
  return r;
}
// B 32x16 f16 (stored transposed [N,K]): lane = 16*g + n ; halves 0..15 -> K = k0+g*16 .. +15
__device__ __forceinline__ v16h load_fragB(const _Float16* __restrict__ p) {
  return *(const v16h*)p;
}

__device__ __forceinline__ float sigmoidf_(float x) { return 1.0f / (1.0f + __expf(-x)); }

// ---------- prep: elementwise f32 -> f16 ----------
__global__ void cvt_f32_f16_k(const float* __restrict__ src, _Float16* __restrict__ dst, int n) {
  int i = blockIdx.x * blockDim.x + threadIdx.x;
  if (i < n) dst[i] = (_Float16)src[i];
}

// ---------- prep: batched transpose+convert  src[b,K,N] f32 -> dst[b,N,K] f16 ----------
__global__ void trb_f32_f16_k(const float* __restrict__ src, _Float16* __restrict__ dst,
                              int K, int N, int total) {
  int i = blockIdx.x * blockDim.x + threadIdx.x;
  if (i >= total) return;
  int n = i % N;
  int k = (i / N) % K;
  int b = i / (N * K);
  dst[(size_t)b * N * K + (size_t)n * K + k] = (_Float16)src[i];
}

// ---------- stage 1: GRU cell (R = h@W_R via WMMA, fused gates) ----------
// 3 independent accumulator chains (u/r/e) per wave hide WMMA latency.
__global__ void __launch_bounds__(128) gru_kernel(
    const _Float16* __restrict__ hPrev16,    // [B,384] f16
    const float*    __restrict__ prev_hidden,// [B,384]
    const float*    __restrict__ prev_y,     // [B,8]
    const float*    __restrict__ cur_c,      // [B,4]
    const _Float16* __restrict__ WtR,        // [1152,384] f16 (W_R transposed)
    const float*    __restrict__ W_Ic,       // [8,576]
    const float*    __restrict__ W_If,       // [12,576]
    const float*    __restrict__ bias_u,
    const float*    __restrict__ bias_r,
    const float*    __restrict__ bias_e,
    float*          __restrict__ hiddenOut,  // [B,384] fp32 (d_out tail)
    _Float16*       __restrict__ hidden16)   // [B,384] f16 (ws)
{
  const int lane    = threadIdx.x & 31;
  const int wave    = threadIdx.x >> 5;
  const int rowBase = (blockIdx.x * 4 + wave) * 16;
  const int colBase = blockIdx.y * 16;
  const int g  = lane >> 4;
  const int mr = lane & 15;

  const _Float16* aRow = hPrev16 + (size_t)(rowBase + mr) * HS + g * 16;
  const _Float16* bU   = WtR + (size_t)(0 * HS + colBase + mr) * HS + g * 16;
  const _Float16* bR   = WtR + (size_t)(1 * HS + colBase + mr) * HS + g * 16;
  const _Float16* bE   = WtR + (size_t)(2 * HS + colBase + mr) * HS + g * 16;

  v8f accU = {}; v8f accR = {}; v8f accE = {};
#pragma unroll
  for (int k0 = 0; k0 < HS; k0 += 32) {
    if (k0 + 32 < HS) __builtin_prefetch(aRow + k0 + 32, 0, 3);
    v16h a  = load_fragA(aRow + k0);
    v16h bu = load_fragB(bU + k0);
    v16h br = load_fragB(bR + k0);
    v16h be = load_fragB(bE + k0);
    accU = __builtin_amdgcn_wmma_f32_16x16x32_f16(false, a, false, bu, (short)0, accU, false, false);
    accR = __builtin_amdgcn_wmma_f32_16x16x32_f16(false, a, false, br, (short)0, accR, false, false);
    accE = __builtin_amdgcn_wmma_f32_16x16x32_f16(false, a, false, be, (short)0, accE, false, false);
  }

  const int n = colBase + mr;                 // hidden column
  const float bu_n = bias_u[n], br_n = bias_r[n], be_n = bias_e[n];
#pragma unroll
  for (int v = 0; v < 8; v++) {
    const int m = rowBase + v + 8 * g;        // batch row (C/D layout)
    const float* py = prev_y + (size_t)m * 8;
    float iu = 0.f, ir = 0.f, ie = 0.f;
    if (n < SS) {                             // coarse part: prev_y @ W_Ic
#pragma unroll
      for (int t = 0; t < 8; t++) {
        float x = py[t];
        iu += x * W_Ic[t * (3 * SS) + n];
        ir += x * W_Ic[t * (3 * SS) + SS + n];
        ie += x * W_Ic[t * (3 * SS) + 2 * SS + n];
      }
    } else {                                  // fine part: [prev_y,cur_c] @ W_If
      const int nf = n - SS;
#pragma unroll
      for (int t = 0; t < 12; t++) {
        float x = (t < 8) ? py[t] : cur_c[(size_t)m * 4 + (t - 8)];
        iu += x * W_If[t * (3 * SS) + nf];
        ir += x * W_If[t * (3 * SS) + SS + nf];
        ie += x * W_If[t * (3 * SS) + 2 * SS + nf];
      }
    }
    float u = sigmoidf_(accU[v] + iu + bu_n);
    float r = sigmoidf_(accR[v] + ir + br_n);
    float e = tanhf(r * accE[v] + ie + be_n);
    float hp = prev_hidden[(size_t)m * HS + n];
    float h  = u * hp + (1.0f - u) * e;
    hiddenOut[(size_t)m * HS + n] = h;
    hidden16[(size_t)m * HS + n]  = (_Float16)h;
  }
}

// ---------- stage 2: out_c / out_f = relu(h_half @ W_O + b) ----------
// 16x32 tile per wave: shared A fragment, 2 independent accumulator chains.
__global__ void __launch_bounds__(128) proj_kernel(
    const _Float16* __restrict__ hidden16,  // [B,384] f16
    const _Float16* __restrict__ WtO1,      // [192,192] f16 (transposed)
    const _Float16* __restrict__ WtO3,
    const float*    __restrict__ b_O1,
    const float*    __restrict__ b_O3,
    _Float16*       __restrict__ outc16,    // [B,192] f16
    _Float16*       __restrict__ outf16)
{
  const int lane    = threadIdx.x & 31;
  const int wave    = threadIdx.x >> 5;
  const int rowBase = (blockIdx.x * 4 + wave) * 16;
  const int colBase = blockIdx.y * 32;
  const int cf      = blockIdx.z;           // 0 = coarse, 1 = fine
  const int g  = lane >> 4;
  const int mr = lane & 15;

  const _Float16* Wt   = cf ? WtO3 : WtO1;
  const float*    bias = cf ? b_O3 : b_O1;
  _Float16*       outp = cf ? outf16 : outc16;

  const _Float16* aRow  = hidden16 + (size_t)(rowBase + mr) * HS + cf * SS + g * 16;
  const _Float16* bRow0 = Wt + (size_t)(colBase + mr) * SS + g * 16;
  const _Float16* bRow1 = Wt + (size_t)(colBase + 16 + mr) * SS + g * 16;

  v8f acc0 = {}; v8f acc1 = {};
#pragma unroll
  for (int k0 = 0; k0 < SS; k0 += 32) {
    v16h a  = load_fragA(aRow + k0);
    v16h b0 = load_fragB(bRow0 + k0);
    v16h b1 = load_fragB(bRow1 + k0);
    acc0 = __builtin_amdgcn_wmma_f32_16x16x32_f16(false, a, false, b0, (short)0, acc0, false, false);
    acc1 = __builtin_amdgcn_wmma_f32_16x16x32_f16(false, a, false, b1, (short)0, acc1, false, false);
  }

  const int n0 = colBase + mr;
  const int n1 = n0 + 16;
  const float bn0 = bias[n0], bn1 = bias[n1];
#pragma unroll
  for (int v = 0; v < 8; v++) {
    const int m = rowBase + v + 8 * g;
    float v0 = acc0[v] + bn0;
    float v1 = acc1[v] + bn1;
    v0 = v0 > 0.f ? v0 : 0.f;
    v1 = v1 > 0.f ? v1 : 0.f;
    outp[(size_t)m * SS + n0] = (_Float16)v0;
    outp[(size_t)m * SS + n1] = (_Float16)v1;
  }
}

// ---------- stage 3: 4 coarse + 4 fine heads -> fp32 logits ----------
// 16x32 tile per wave: shared A fragment, 2 independent accumulator chains.
__global__ void __launch_bounds__(128) head_kernel(
    const _Float16* __restrict__ outc16,    // [B,192] f16
    const _Float16* __restrict__ outf16,
    const _Float16* __restrict__ Wt2c,      // [4,256,192] f16 (per-head transposed)
    const _Float16* __restrict__ Wt2f,
    const float*    __restrict__ b2c,       // [4,256]
    const float*    __restrict__ b2f,
    float*          __restrict__ out,       // d_out: c[4B,Q] then f[4B,Q]
    int B)
{
  const int lane    = threadIdx.x & 31;
  const int wave    = threadIdx.x >> 5;
  const int rowBase = (blockIdx.x * 4 + wave) * 16;
  const int colBase = blockIdx.y * 32;
  const int z    = blockIdx.z;
  const int cf   = z >> 2;                  // 0 = coarse heads, 1 = fine heads
  const int head = z & 3;
  const int g  = lane >> 4;
  const int mr = lane & 15;

  const _Float16* A    = cf ? outf16 : outc16;
  const _Float16* Wt   = (cf ? Wt2f : Wt2c) + (size_t)head * QQ * SS;
  const float*    bias = (cf ? b2f : b2c) + head * QQ;
  float* dst = out + (size_t)cf * 4 * B * QQ + (size_t)head * B * QQ;

  const _Float16* aRow  = A + (size_t)(rowBase + mr) * SS + g * 16;
  const _Float16* bRow0 = Wt + (size_t)(colBase + mr) * SS + g * 16;
  const _Float16* bRow1 = Wt + (size_t)(colBase + 16 + mr) * SS + g * 16;

  v8f acc0 = {}; v8f acc1 = {};
#pragma unroll
  for (int k0 = 0; k0 < SS; k0 += 32) {
    if (k0 + 32 < SS) __builtin_prefetch(aRow + k0 + 32, 0, 3);
    v16h a  = load_fragA(aRow + k0);
    v16h b0 = load_fragB(bRow0 + k0);
    v16h b1 = load_fragB(bRow1 + k0);
    acc0 = __builtin_amdgcn_wmma_f32_16x16x32_f16(false, a, false, b0, (short)0, acc0, false, false);
    acc1 = __builtin_amdgcn_wmma_f32_16x16x32_f16(false, a, false, b1, (short)0, acc1, false, false);
  }

  const int n0 = colBase + mr;
  const int n1 = n0 + 16;
  const float bn0 = bias[n0], bn1 = bias[n1];
#pragma unroll
  for (int v = 0; v < 8; v++) {
    const int m = rowBase + v + 8 * g;
    dst[(size_t)m * QQ + n0] = acc0[v] + bn0;
    dst[(size_t)m * QQ + n1] = acc1[v] + bn1;
  }
}

extern "C" void kernel_launch(void* const* d_in, const int* in_sizes, int n_in,
                              void* d_out, int out_size, void* d_ws, size_t ws_size,
                              hipStream_t stream) {
  const float* prev_y      = (const float*)d_in[0];
  const float* prev_hidden = (const float*)d_in[1];
  const float* cur_c       = (const float*)d_in[2];
  const float* W_R         = (const float*)d_in[3];
  const float* W_Ic        = (const float*)d_in[4];
  const float* W_If        = (const float*)d_in[5];
  const float* W_O1        = (const float*)d_in[6];
  const float* b_O1        = (const float*)d_in[7];
  const float* W_O3        = (const float*)d_in[8];
  const float* b_O3        = (const float*)d_in[9];
  const float* W2c         = (const float*)d_in[10];
  const float* b2c         = (const float*)d_in[11];
  const float* W2f         = (const float*)d_in[12];
  const float* b2f         = (const float*)d_in[13];
  const float* bias_u      = (const float*)d_in[14];
  const float* bias_r      = (const float*)d_in[15];
  const float* bias_e      = (const float*)d_in[16];
  float* out = (float*)d_out;

  const int B = in_sizes[0] / 8;   // 16384

  // ---- workspace layout (f16 activations + transposed f16 weights) ----
  char* ws = (char*)d_ws;
  size_t o = 0;
  _Float16* hPrev16 = (_Float16*)(ws + o); o += (size_t)B * HS * 2;
  _Float16* hNew16  = (_Float16*)(ws + o); o += (size_t)B * HS * 2;
  _Float16* outc16  = (_Float16*)(ws + o); o += (size_t)B * SS * 2;
  _Float16* outf16  = (_Float16*)(ws + o); o += (size_t)B * SS * 2;
  _Float16* WtR     = (_Float16*)(ws + o); o += (size_t)3 * HS * HS * 2;
  _Float16* WtO1    = (_Float16*)(ws + o); o += (size_t)SS * SS * 2;
  _Float16* WtO3    = (_Float16*)(ws + o); o += (size_t)SS * SS * 2;
  _Float16* Wt2c    = (_Float16*)(ws + o); o += (size_t)4 * QQ * SS * 2;
  _Float16* Wt2f    = (_Float16*)(ws + o); o += (size_t)4 * QQ * SS * 2;

  // ---- prep: convert / transpose ----
  {
    int n = B * HS;
    cvt_f32_f16_k<<<(n + 255) / 256, 256, 0, stream>>>(prev_hidden, hPrev16, n);
  }
  {
    int t = HS * 3 * HS;  // W_R [384,1152] -> [1152,384]
    trb_f32_f16_k<<<(t + 255) / 256, 256, 0, stream>>>(W_R, WtR, HS, 3 * HS, t);
  }
  {
    int t = SS * SS;
    trb_f32_f16_k<<<(t + 255) / 256, 256, 0, stream>>>(W_O1, WtO1, SS, SS, t);
    trb_f32_f16_k<<<(t + 255) / 256, 256, 0, stream>>>(W_O3, WtO3, SS, SS, t);
  }
  {
    int t = 4 * SS * QQ;  // [4,192,256] -> [4,256,192]
    trb_f32_f16_k<<<(t + 255) / 256, 256, 0, stream>>>(W2c, Wt2c, SS, QQ, t);
    trb_f32_f16_k<<<(t + 255) / 256, 256, 0, stream>>>(W2f, Wt2f, SS, QQ, t);
  }

  float* hiddenOut = out + (size_t)8 * B * QQ;  // after c[4B,Q] and f[4B,Q]
  dim3 blk(128);

  // ---- stage 1: GRU ----
  gru_kernel<<<dim3(B / 64, HS / 16), blk, 0, stream>>>(
      hPrev16, prev_hidden, prev_y, cur_c, WtR, W_Ic, W_If,
      bias_u, bias_r, bias_e, hiddenOut, hNew16);

  // ---- stage 2: O1/O3 projections (16x32 tiles) ----
  proj_kernel<<<dim3(B / 64, SS / 32, 2), blk, 0, stream>>>(
      hNew16, WtO1, WtO3, b_O1, b_O3, outc16, outf16);

  // ---- stage 3: 8 output heads (16x32 tiles) ----
  head_kernel<<<dim3(B / 64, QQ / 32, 8), blk, 0, stream>>>(
      outc16, outf16, Wt2c, Wt2f, b2c, b2f, out, B);
}
